// JointStep_53704271069292
// MI455X (gfx1250) — matrix operand order repaired and checked by reading, not compiled, not run on hardware
//
#include <hip/hip_runtime.h>

typedef _Float16 half_t;
typedef __attribute__((ext_vector_type(16))) _Float16 v16h;
typedef __attribute__((ext_vector_type(8)))  float    v8f;

namespace {
constexpr int   kNTok    = 4096;   // tokens
constexpr int   kVocab   = 4096;   // visual codes (8*8*8*8)
constexpr int   kFullV   = 4098;   // + ALIVE/DEATH
constexpr int   kRows    = 16;     // tokens per workgroup == WMMA M/N
constexpr int   kThreads = 256;    // 8 x wave32
constexpr int   kChunks  = kVocab / 32;  // 128 K-chunks for 16x16x32 WMMA
constexpr int   kPStride = 68;     // padded row stride for weight tables (bank spread)
constexpr float kInv2Sig2 = 1.0f / (2.0f * 0.9f * 0.9f);   // SIGMA = 0.9
constexpr float kSmooth   = 0.1f;                          // SMOOTHING
}

__global__ __launch_bounds__(kThreads)
void fsq_focal_rows(const float* __restrict__ logits,
                    const float* __restrict__ gamma,
                    const int*   __restrict__ tgt,
                    float* __restrict__ focal_out)
{
  __shared__ __align__(16) float sP01[kRows * kPStride];  // E0*E1 pairwise table per row
  __shared__ __align__(16) float sP23[kRows * kPStride];  // E2*E3 pairwise table per row
  __shared__ float sE[32];          // E_d[delta], d=0..3, delta=0..7
  __shared__ float sPartM[kThreads];
  __shared__ float sPartS[kThreads];
  __shared__ float sRowMax[kRows];
  __shared__ float sRowKf[kRows];   // closed-form full kernel row sum
  __shared__ float sRowXt[kRows];   // logits[row, t]
  __shared__ int   sRowT[kRows];
  __shared__ int   sRowDig[kRows * 4];
  __shared__ float sDiag[8 * kRows];

  const int tid = threadIdx.x;
  const int r0  = blockIdx.x * kRows;

  // ---- per-dimension Gaussian tables: E_d[delta] = exp(-(w_d*delta)^2/(2*sigma^2))
  if (tid < 32) {
    const int d = tid >> 3, dl = tid & 7;
    const float w = __expf(gamma[d]);
    const float t = w * (float)dl;
    sE[tid] = __expf(-(t * t) * kInv2Sig2);
  }
  __syncthreads();

  // ---- per-row target digits, x[t], closed-form kernel sum Kfull = prod_d S_d
  if (tid < kRows) {
    const int t    = tgt[r0 + tid];
    const int safe = t < (kVocab - 1) ? t : (kVocab - 1);
    const int dig[4] = { (safe >> 9) & 7, (safe >> 6) & 7, (safe >> 3) & 7, safe & 7 };
    #pragma unroll
    for (int d = 0; d < 4; ++d) sRowDig[tid * 4 + d] = dig[d];
    float kf = 1.0f;
    #pragma unroll
    for (int d = 0; d < 4; ++d) {
      float ssum = 0.0f;
      #pragma unroll
      for (int c = 0; c < 8; ++c) {
        const int delta = c > dig[d] ? c - dig[d] : dig[d] - c;
        ssum += sE[d * 8 + delta];
      }
      kf *= ssum;
    }
    sRowKf[tid] = kf;
    sRowT[tid]  = t;
    sRowXt[tid] = logits[(size_t)(r0 + tid) * kFullV + t];
  }
  __syncthreads();

  // ---- pairwise weight tables: P01[r][c0*8+c1], P23[r][c2*8+c3]
  for (int e = tid; e < kRows * 128; e += kThreads) {
    const int r = e >> 7;
    const int k = e & 127;
    const int hi6 = (k & 64) ? 2 : 0;              // selects dims (2,3) vs (0,1)
    const int ca = (k >> 3) & 7, cb_ = k & 7;
    const int da = sRowDig[r * 4 + hi6], db = sRowDig[r * 4 + hi6 + 1];
    const int aa = ca > da ? ca - da : da - ca;
    const int ab = cb_ > db ? cb_ - db : db - cb_;
    const float val = sE[hi6 * 8 + aa] * sE[(hi6 + 1) * 8 + ab];
    if (k < 64) sP01[r * kPStride + k]        = val;
    else        sP23[r * kPStride + (k - 64)] = val;
  }
  __syncthreads();

  const int wave = tid >> 5;
  const int lane = tid & 31;
  const int arow = lane & 15;   // token row inside tile (= A matrix M, = B matrix N)
  const int hi   = lane >> 4;   // K-group selector per ISA 16-bit A/B layouts
  const float* __restrict__ rp = logits + (size_t)(r0 + arow) * kFullV;

  // ================= PASS A: row max (pure fmax streaming read) ==============
  float m = (wave == 0) ? rp[kVocab + hi] : -__builtin_inff();
  for (int q = wave; q < kChunks; q += 8) {
    const int cb = q << 5;
    float2 xv[8];
    #pragma unroll
    for (int i = 0; i < 4; ++i) xv[i]     = *(const float2*)(rp + cb + hi * 8 + 2 * i);
    #pragma unroll
    for (int i = 0; i < 4; ++i) xv[4 + i] = *(const float2*)(rp + cb + 16 + hi * 8 + 2 * i);
    #pragma unroll
    for (int i = 0; i < 8; ++i) m = fmaxf(m, fmaxf(xv[i].x, xv[i].y));
  }
  sPartM[tid] = m;
  __syncthreads();
  if (tid < kRows) {
    float mm = -__builtin_inff();
    #pragma unroll
    for (int w = 0; w < 8; ++w) {
      mm = fmaxf(mm, sPartM[w * 32 + tid]);
      mm = fmaxf(mm, sPartM[w * 32 + 16 + tid]);
    }
    sRowMax[tid] = mm;
  }
  __syncthreads();

  // ===== PASS B: exp-sum (branch-free, exact max) + f16 A-frag + WMMA ========
  const float rmax = sRowMax[arow];
  float s = (wave == 0) ? __expf(rp[kVocab + hi] - rmax) : 0.0f;

  const float* __restrict__ p01 = sP01 + arow * kPStride;
  const float* __restrict__ p23 = sP23 + arow * kPStride;
  const int kb = hi << 4;

  v8f acc = {0.f, 0.f, 0.f, 0.f, 0.f, 0.f, 0.f, 0.f};

  for (int q = wave; q < kChunks; q += 8) {
    const int cb = q << 5;

    // batched loads (L2-resident after pass A) for the A fragment:
    // halves 0..15 -> K = cb + 16*hi + 0..15
    float2 xv[8];
    #pragma unroll
    for (int i = 0; i < 8; ++i) xv[i] = *(const float2*)(rp + cb + hi * 16 + 2 * i);

    v16h A;
    #pragma unroll
    for (int i = 0; i < 8; ++i) {
      s += __expf(xv[i].x - rmax);
      s += __expf(xv[i].y - rmax);
      A[2 * i]     = (half_t)xv[i].x;
      A[2 * i + 1] = (half_t)xv[i].y;
    }

    // B fragment (16-bit B 32x16 layout): col = arow, half h -> K = kb + h.
    // Within a 32-wide chunk the (c0,c1) digit pair is constant -> one P01 scalar;
    // (c2,c3) spans 16 consecutive entries of P23.
    const float w01 = p01[(cb + kb) >> 6];
    const int   b23 = (cb & 63) + kb;
    const float4 qa = *(const float4*)(p23 + b23);
    const float4 qb = *(const float4*)(p23 + b23 + 4);
    const float4 qc = *(const float4*)(p23 + b23 + 8);
    const float4 qd = *(const float4*)(p23 + b23 + 12);
    v16h B;
    B[0]  = (half_t)(w01 * qa.x);  B[1]  = (half_t)(w01 * qa.y);
    B[2]  = (half_t)(w01 * qa.z);  B[3]  = (half_t)(w01 * qa.w);
    B[4]  = (half_t)(w01 * qb.x);  B[5]  = (half_t)(w01 * qb.y);
    B[6]  = (half_t)(w01 * qb.z);  B[7]  = (half_t)(w01 * qb.w);
    B[8]  = (half_t)(w01 * qc.x);  B[9]  = (half_t)(w01 * qc.y);
    B[10] = (half_t)(w01 * qc.z);  B[11] = (half_t)(w01 * qc.w);
    B[12] = (half_t)(w01 * qd.x);  B[13] = (half_t)(w01 * qd.y);
    B[14] = (half_t)(w01 * qd.z);  B[15] = (half_t)(w01 * qd.w);

    // C[r][j] += sum_k X[r][k] * W_j[k]; we need only the diagonal r==j.
    acc = __builtin_amdgcn_wmma_f32_16x16x32_f16(
        /*neg_a=*/false, A, /*neg_b=*/false, B,
        /*c_mod=*/(short)0, acc, /*reuse_a=*/false, /*reuse_b=*/false);
  }

  sPartS[tid] = s;

  // diagonal of 16x16 f32 C: lanes 0-7 hold rows 0-7 (acc[lane]),
  // lanes 24-31 hold rows 8-15 (acc[lane-24])
  int sel = -1, drow = -1;
  if (lane < 8)        { sel = lane;      drow = lane; }
  else if (lane >= 24) { sel = lane - 24; drow = lane - 16; }
  float dv = 0.0f;
  #pragma unroll
  for (int i = 0; i < 8; ++i) if (sel == i) dv = acc[i];
  if (drow >= 0) sDiag[wave * kRows + drow] = dv;
  __syncthreads();

  // ---- per-row finalize: combine stats, assemble ce, focal term
  if (tid < kRows) {
    const int r = tid;
    float zz = 0.0f;
    #pragma unroll
    for (int w = 0; w < 8; ++w) {
      zz += sPartS[w * 32 + r];
      zz += sPartS[w * 32 + 16 + r];
    }
    const float mz = sRowMax[r] + __logf(zz);   // m + logZ

    float skx = 0.0f;                   // sum_v kernel(v,t) * x[v] (unmasked)
    #pragma unroll
    for (int w = 0; w < 8; ++w) skx += sDiag[w * kRows + r];

    const int   t  = sRowT[r];
    const float xt = sRowXt[r];
    float dot;
    if (t < kVocab) {
      const float rowsum = fmaxf(sRowKf[r] - 1.0f, 1e-8f);
      dot = (1.0f - kSmooth) * xt + (kSmooth / rowsum) * (skx - xt);
    } else {
      dot = xt;                         // status token: one-hot target
    }
    const float ce = mz - dot;
    const float pt = __expf(-ce);
    const float om = 1.0f - pt;
    focal_out[r0 + r] = om * om * ce;   // FOCAL_GAMMA = 2
  }
}

__global__ __launch_bounds__(256)
void fsq_focal_reduce(const float* __restrict__ focal, float* __restrict__ out)
{
  __shared__ float red[256];
  float s = 0.0f;
  for (int i = threadIdx.x; i < kNTok; i += 256) s += focal[i];
  red[threadIdx.x] = s;
  __syncthreads();
  #pragma unroll
  for (int off = 128; off > 0; off >>= 1) {
    if (threadIdx.x < off) red[threadIdx.x] += red[threadIdx.x + off];
    __syncthreads();
  }
  if (threadIdx.x == 0) out[0] = red[0] / (float)kNTok;
}

extern "C" void kernel_launch(void* const* d_in, const int* in_sizes, int n_in,
                              void* d_out, int out_size, void* d_ws, size_t ws_size,
                              hipStream_t stream) {
  (void)in_sizes; (void)n_in; (void)out_size; (void)ws_size;
  const float* logits = (const float*)d_in[0];
  // d_in[1] = coords: mixed-radix grid, regenerated analytically on device
  const float* gamma  = (const float*)d_in[2];
  const int*   tgt    = (const int*)d_in[3];
  float* focal = (float*)d_ws;               // 4096 floats of scratch

  fsq_focal_rows<<<dim3(kNTok / kRows), dim3(kThreads), 0, stream>>>(
      logits, gamma, tgt, focal);
  fsq_focal_reduce<<<dim3(1), dim3(256), 0, stream>>>(focal, (float*)d_out);
}
